// Attention_37194416783883
// MI455X (gfx1250) — matrix operand order repaired and checked by reading, not compiled, not run on hardware
//
#include <hip/hip_runtime.h>
#include <hip/hip_fp16.h>
#include <cstdint>

// ---------------------------------------------------------------------------
// MHA forward, fused flash-attention style for MI455X (gfx1250).
// All GEMM stages: v_wmma_f32_16x16x32_f16 (f32 accum).
// K/V tiles double-buffered into LDS with global_load_async_to_lds_b128.
// B=4, N=2048, DIM=64, H=8 -> M = 8192 tokens.
// ---------------------------------------------------------------------------

typedef _Float16 h16 __attribute__((ext_vector_type(16)));
typedef float    f8  __attribute__((ext_vector_type(8)));

union H16 { h16 v; uint4 q[2]; _Float16 e[16]; };

#define QK_SCALE 0.125f   // 64^-0.5

static constexpr int NB   = 4;
static constexpr int NN   = 2048;
static constexpr int DIM  = 64;
static constexpr int NH   = 8;
static constexpr int QKVW = 3 * NH * DIM;   // 1536
static constexpr int HD   = NH * DIM;       // 512
static constexpr int MTOT = NB * NN;        // 8192
static constexpr int KVT  = 64;             // kv tile width

// WMMA operand layouts as used in this file (wave32, contiguous k order in A and B):
//  A (16x32 f16): row m = lane&15; kbA = 16*(lane>=16); elem i -> K=kbA+i
//  B (32x16 f16): col n = lane&15; kbB = 16*(lane>=16); elem i -> K=kbB+i
//  C/D (16x16 f32): elem r -> row = 8*(lane>=16)+r, col = lane&15

// CDNA5 async copy: global -> LDS, tracked by ASYNCcnt (ISA 10.x / 15.18.3)
__device__ inline void async_b128(uint32_t lds_byte, const void* gptr) {
    asm volatile("global_load_async_to_lds_b128 %0, %1, off"
                 :: "v"(lds_byte), "v"(gptr) : "memory");
}
__device__ inline uint32_t lds_addr(const void* p) {
    // generic LDS pointer: low 32 bits are the wave-relative LDS byte address
    return (uint32_t)(uintptr_t)p;
}

// ---------------------------------------------------------------------------
// Stage 0: transpose weights to f16, [n][k] layout -> contiguous WMMA B tiles
// ---------------------------------------------------------------------------
__global__ __launch_bounds__(256) void wt_kernel(
    const float* __restrict__ Wqkv, const float* __restrict__ Wproj,
    _Float16* __restrict__ Wqkvt, _Float16* __restrict__ Wpt)
{
    const int id = blockIdx.x * 256 + threadIdx.x;
    if (id < QKVW * DIM) {                       // Wqkvt[n][k], K=64, n<1536
        const int n = id >> 6, k = id & 63;
        Wqkvt[id] = (_Float16)Wqkv[(size_t)k * QKVW + n];
    }
    if (id < DIM * HD) {                         // Wpt[n][k], K=512, n<64
        const int n = id >> 9, k = id & 511;
        Wpt[id] = (_Float16)Wproj[(size_t)k * DIM + n];
    }
}

// ---------------------------------------------------------------------------
// Stage 1: qkv = x @ Wqkv ; scatter to Qh/Kh [B,H,N,64] f16 and Vt [B,H,64,N]
// ---------------------------------------------------------------------------
__global__ __launch_bounds__(128) void qkv_kernel(
    const float* __restrict__ x, const _Float16* __restrict__ Wqkvt,
    _Float16* __restrict__ Qh, _Float16* __restrict__ Kh,
    _Float16* __restrict__ Vt)
{
    const int lane = threadIdx.x & 31;
    const int wid  = threadIdx.x >> 5;
    const int n0   = (blockIdx.x * 4 + wid) * 16;
    const int m0   = blockIdx.y * 16;
    const int col  = lane & 15;
    const int hi   = lane >> 4;
    const int kbA  = hi * 16;
    const int kbB  = hi * 16;

    f8 acc = {};
    #pragma unroll
    for (int cb = 0; cb < DIM; cb += 32) {
        H16 a, b;
        const float* ax = x + (size_t)(m0 + col) * DIM + cb + kbA;
        #pragma unroll
        for (int i = 0; i < 8; ++i) a.e[i]     = (_Float16)ax[i];
        #pragma unroll
        for (int i = 0; i < 8; ++i) a.e[8 + i] = (_Float16)ax[8 + i];
        const _Float16* bw = Wqkvt + (size_t)(n0 + col) * DIM + cb + kbB;
        b.q[0] = *(const uint4*)bw;
        b.q[1] = *(const uint4*)(bw + 8);
        acc = __builtin_amdgcn_wmma_f32_16x16x32_f16(false, a.v, false, b.v,
                                                     (short)0, acc, false, false);
    }

    const int ng  = n0 + col;
    const int sel = ng >> 9;          // 0=q 1=k 2=v
    const int h   = (ng >> 6) & 7;
    const int c   = ng & 63;
    const int b_  = m0 >> 11;
    const int tok = (m0 & (NN - 1)) + hi * 8;
    const size_t bh = (size_t)(b_ * NH + h);

    union { _Float16 e[8]; uint4 q; } ov;
    #pragma unroll
    for (int r = 0; r < 8; ++r) ov.e[r] = (_Float16)acc[r];

    if (sel == 2) {      // V stored transposed: 8 consecutive tokens -> 1 b128
        *(uint4*)&Vt[(bh * DIM + c) * NN + tok] = ov.q;
    } else {
        _Float16* dst = (sel ? Kh : Qh) + (bh * NN + tok) * DIM + c;
        #pragma unroll
        for (int r = 0; r < 8; ++r) dst[(size_t)r * DIM] = ov.e[r];
    }
}

// ---------------------------------------------------------------------------
// Stage 2: flash attention. Block = 4 waves x 16 query rows = 64 rows.
// 64-wide KV tiles staged to LDS with async copies, double-buffered.
// ---------------------------------------------------------------------------
__global__ __launch_bounds__(128) void attn_kernel(
    const _Float16* __restrict__ Qh, const _Float16* __restrict__ Kh,
    const _Float16* __restrict__ Vt, _Float16* __restrict__ AO)
{
    __shared__ __align__(16) _Float16 Kl[2][KVT * DIM];   // [buf][token][c] 8KB
    __shared__ __align__(16) _Float16 Vl[2][DIM * KVT];   // [buf][c][token] 8KB
    __shared__ __align__(16) _Float16 Pl[4][16][KVT];     // per-wave P tile

    const int tid  = threadIdx.x;
    const int lane = tid & 31;
    const int wid  = tid >> 5;
    const int bh   = blockIdx.x;                   // b*8+h
    const int b_   = bh >> 3, h = bh & 7;
    const int m0   = blockIdx.y * 64 + wid * 16;
    const int col  = lane & 15;
    const int hi   = lane >> 4;
    const int kbA  = hi * 16;
    const int kbB  = hi * 16;

    const _Float16* Qb = Qh + (size_t)bh * NN * DIM;
    const _Float16* Kb = Kh + (size_t)bh * NN * DIM;
    const _Float16* Vb = Vt + (size_t)bh * DIM * NN;

    // cooperative async stage of one 64-wide KV tile into LDS buffer `buf`
    auto stage = [&](int kv, int buf) {
        const uint32_t kbase = lds_addr(&Kl[buf][0]);
        const uint32_t vbase = lds_addr(&Vl[buf][0]);
        const char* kg = (const char*)(Kb + (size_t)kv * DIM);  // 8KB contiguous
        #pragma unroll
        for (int j = 0; j < 4; ++j) {
            const uint32_t off = (uint32_t)(tid + j * 128) * 16;
            async_b128(kbase + off, kg + off);
        }
        #pragma unroll
        for (int j = 0; j < 4; ++j) {               // V: 64 rows of 128B
            const uint32_t chunk = (uint32_t)(tid + j * 128);
            const uint32_t row   = chunk >> 3;      // feature c
            const uint32_t coff  = (chunk & 7) * 16;
            const char* vg = (const char*)(Vb + (size_t)row * NN + kv) + coff;
            async_b128(vbase + row * 128 + coff, vg);
        }
    };

    // Q tile in A-layout (both 32-wide chunks), resident in VGPRs
    H16 qa[2];
    {
        const _Float16* qp = Qb + (size_t)(m0 + col) * DIM;
        #pragma unroll
        for (int cc = 0; cc < 2; ++cc) {
            const _Float16* p = qp + cc * 32 + kbA;
            qa[cc].q[0] = *(const uint4*)p;
            qa[cc].q[1] = *(const uint4*)(p + 8);
        }
    }

    f8 O[4] = {};
    float rm[8], rl[8];
    #pragma unroll
    for (int r = 0; r < 8; ++r) { rm[r] = -1e30f; rl[r] = 0.f; }

    stage(0, 0);
    for (int t = 0; t < NN / KVT; ++t) {
        const int buf = t & 1;
        if (t + 1 < NN / KVT) {
            stage((t + 1) * KVT, buf ^ 1);          // prefetch next tile
            asm volatile("s_wait_asynccnt 0x8" ::: "memory");  // tile t landed
        } else {
            asm volatile("s_wait_asynccnt 0x0" ::: "memory");
        }
        __syncthreads();

        const _Float16* Kt = &Kl[buf][0];
        const _Float16* Vs = &Vl[buf][0];

        // ---- S = Q @ K^T : 4 column tiles x 2 K-chunks = 8 WMMAs ----
        f8 S[4] = {};
        #pragma unroll
        for (int cc = 0; cc < 2; ++cc)
            #pragma unroll
            for (int nt = 0; nt < 4; ++nt) {
                H16 bmat;
                const _Float16* p = Kt + (size_t)(nt * 16 + col) * DIM + cc * 32 + kbB;
                bmat.q[0] = *(const uint4*)p;
                bmat.q[1] = *(const uint4*)(p + 8);
                S[nt] = __builtin_amdgcn_wmma_f32_16x16x32_f16(
                    false, qa[cc].v, false, bmat.v, (short)0, S[nt], false, false);
            }

        // ---- online softmax over 64 columns (xor 1/2/4/8 == row reduce) ----
        float alpha[8];
        #pragma unroll
        for (int r = 0; r < 8; ++r) {
            float mx = fmaxf(fmaxf(S[0][r], S[1][r]),
                             fmaxf(S[2][r], S[3][r])) * QK_SCALE;
            #pragma unroll
            for (int off = 1; off < 16; off <<= 1)
                mx = fmaxf(mx, __shfl_xor(mx, off, 32));
            const float mnew = fmaxf(rm[r], mx);
            alpha[r] = __expf(rm[r] - mnew);
            float p0 = __expf(S[0][r] * QK_SCALE - mnew);
            float p1 = __expf(S[1][r] * QK_SCALE - mnew);
            float p2 = __expf(S[2][r] * QK_SCALE - mnew);
            float p3 = __expf(S[3][r] * QK_SCALE - mnew);
            float rs = (p0 + p1) + (p2 + p3);
            #pragma unroll
            for (int off = 1; off < 16; off <<= 1)
                rs += __shfl_xor(rs, off, 32);
            rl[r] = rl[r] * alpha[r] + rs;
            rm[r] = mnew;
            Pl[wid][hi * 8 + r][col]      = (_Float16)p0;   // D-layout -> LDS
            Pl[wid][hi * 8 + r][col + 16] = (_Float16)p1;
            Pl[wid][hi * 8 + r][col + 32] = (_Float16)p2;
            Pl[wid][hi * 8 + r][col + 48] = (_Float16)p3;
        }
        #pragma unroll
        for (int ct = 0; ct < 4; ++ct)
            #pragma unroll
            for (int r = 0; r < 8; ++r) O[ct][r] *= alpha[r];

        // same-wave LDS RAW: DS ops complete wave-wide, only need dscnt
        asm volatile("s_wait_dscnt 0x0" ::: "memory");

        // P back in A-layout, both 32-wide chunks
        H16 pa[2];
        #pragma unroll
        for (int pc = 0; pc < 2; ++pc) {
            const _Float16* p = &Pl[wid][col][pc * 32 + kbA];
            pa[pc].q[0] = *(const uint4*)p;
            pa[pc].q[1] = *(const uint4*)(p + 8);
        }

        // ---- O += P @ V : 4 c tiles x 2 chunks = 8 WMMAs ----
        #pragma unroll
        for (int ct = 0; ct < 4; ++ct)
            #pragma unroll
            for (int pc = 0; pc < 2; ++pc) {
                H16 vb;
                const _Float16* vp = Vs + (size_t)(ct * 16 + col) * KVT + pc * 32 + kbB;
                vb.q[0] = *(const uint4*)vp;
                vb.q[1] = *(const uint4*)(vp + 8);
                O[ct] = __builtin_amdgcn_wmma_f32_16x16x32_f16(
                    false, pa[pc].v, false, vb.v, (short)0, O[ct], false, false);
            }
        __syncthreads();   // release buf for the next prefetch
    }

    // normalize, store AO[b, tok, h*64 + c] f16 for the out-projection
    #pragma unroll
    for (int ct = 0; ct < 4; ++ct)
        #pragma unroll
        for (int r = 0; r < 8; ++r) {
            const int tok = m0 + hi * 8 + r;
            const float v = O[ct][r] / rl[r];
            AO[((size_t)b_ * NN + tok) * HD + h * DIM + ct * 16 + col] = (_Float16)v;
        }
}

// ---------------------------------------------------------------------------
// Stage 3: out = AO @ Wproj + bproj  (K = 512), fp32 output
// ---------------------------------------------------------------------------
__global__ __launch_bounds__(128) void proj_kernel(
    const _Float16* __restrict__ AO, const _Float16* __restrict__ Wpt,
    const float* __restrict__ bproj, float* __restrict__ out)
{
    const int lane = threadIdx.x & 31;
    const int wid  = threadIdx.x >> 5;
    const int m0   = blockIdx.x * 16;
    const int n0   = wid * 16;
    const int col  = lane & 15;
    const int hi   = lane >> 4;
    const int kbA  = hi * 16;
    const int kbB  = hi * 16;

    f8 acc = {};
    const _Float16* arow = AO + (size_t)(m0 + col) * HD;
    const _Float16* brow = Wpt + (size_t)(n0 + col) * HD;
    #pragma unroll 4
    for (int cb = 0; cb < HD; cb += 32) {
        H16 a, b;
        const _Float16* p = arow + cb + kbA;
        a.q[0] = *(const uint4*)p;
        a.q[1] = *(const uint4*)(p + 8);
        const _Float16* w = brow + cb + kbB;
        b.q[0] = *(const uint4*)w;
        b.q[1] = *(const uint4*)(w + 8);
        acc = __builtin_amdgcn_wmma_f32_16x16x32_f16(false, a.v, false, b.v,
                                                     (short)0, acc, false, false);
    }
    const float bias = bproj[n0 + col];
    #pragma unroll
    for (int r = 0; r < 8; ++r) {
        const int m = m0 + hi * 8 + r;
        out[(size_t)m * DIM + n0 + col] = acc[r] + bias;
    }
}

// ---------------------------------------------------------------------------
extern "C" void kernel_launch(void* const* d_in, const int* in_sizes, int n_in,
                              void* d_out, int out_size, void* d_ws, size_t ws_size,
                              hipStream_t stream)
{
    const float* x     = (const float*)d_in[0];
    const float* Wqkv  = (const float*)d_in[1];
    const float* Wproj = (const float*)d_in[2];
    const float* bproj = (const float*)d_in[3];
    float* out = (float*)d_out;

    const size_t qkv_elems = (size_t)NB * NH * NN * DIM;   // 4,194,304
    _Float16* Qh    = (_Float16*)d_ws;
    _Float16* Kh    = Qh + qkv_elems;
    _Float16* Vt    = Kh + qkv_elems;
    _Float16* AO    = Vt + qkv_elems;                       // [B,N,512]
    _Float16* Wqkvt = AO + qkv_elems;                       // [1536][64]
    _Float16* Wpt   = Wqkvt + (size_t)QKVW * DIM;           // [64][512]

    wt_kernel<<<(QKVW * DIM + 255) / 256, 256, 0, stream>>>(Wqkv, Wproj, Wqkvt, Wpt);
    qkv_kernel<<<dim3(QKVW / 64, MTOT / 16), 128, 0, stream>>>(x, Wqkvt, Qh, Kh, Vt);
    attn_kernel<<<dim3(NB * NH, NN / 64), 128, 0, stream>>>(Qh, Kh, Vt, AO);
    proj_kernel<<<MTOT / 16, 128, 0, stream>>>(AO, Wpt, bproj, out);
}
